// GT_Model_69801808494862
// MI455X (gfx1250) — matrix operand order repaired and checked by reading, not compiled, not run on hardware
//
#include <hip/hip_runtime.h>
#include <hip/hip_bf16.h>

// ---------------------------------------------------------------------------
// Graph-Transformer forward for MI455X (gfx1250, wave32, WMMA).
// Dense projections (QKV+skip, edge embed) run on v_wmma_f32_16x16x32_f16.
// v2: QKV GEMM stages the 32x128 activation tile in LDS (A frags via
// ds_load_b128, 4x less global A traffic) and preloads all 8 B frags of a
// k-step in one clause so WMMAs issue back-to-back instead of stalling on
// s_wait_loadcnt 0 per tile.  Edge GEMM keeps its 8 B frags register-resident
// and sweeps 8 row-tiles per wave to amortize them.
// ---------------------------------------------------------------------------

#define NN   50000
#define EE   400000
#define GG   128
#define HC   128
#define NH   4
#define CH   32
#define ED   13

typedef __attribute__((ext_vector_type(16))) _Float16 v16h;
typedef __attribute__((ext_vector_type(8)))  float    v8f;

union AFrag { v16h v; uint4 q[2]; unsigned u[8]; _Float16 h[16]; };

// ---------------------------------------------------------------- utilities
__global__ void k_zero(float* p, long n) {
    long i = (long)blockIdx.x * blockDim.x + threadIdx.x;
    if (i < n) p[i] = 0.f;
}

__global__ void k_cvt_f16(const float* __restrict__ src, _Float16* __restrict__ dst, long n) {
    long i = (long)blockIdx.x * blockDim.x + threadIdx.x;
    if (i < n) dst[i] = (_Float16)src[i];
}

// ------------------------------------------------------- weight pre-packing
// Pack [Wq|Wk|Wv|Wskip] (each 128x128, row-major KxN) into WMMA B-fragment
// layout: frag (kt,nt), lane l (n=l&15, hh=l>>4), dword r holds
// (W[k][n], W[k+1][n]) with k = kt*32 + hh*16 + 2r.
__global__ void k_pack_qkvs(const float* __restrict__ Wq, const float* __restrict__ Wk,
                            const float* __restrict__ Wv, const float* __restrict__ Ws,
                            unsigned* __restrict__ pk) {
    int tile = blockIdx.x;          // 0..127 = kt*32 + nt
    int lane = threadIdx.x;         // 32 lanes
    int kt = tile >> 5, nt = tile & 31;
    int nl = lane & 15, hh = lane >> 4;
    const float* W = (nt < 8) ? Wq : (nt < 16) ? Wk : (nt < 24) ? Wv : Ws;
    int col = (nt & 7) * 16 + nl;
    #pragma unroll
    for (int r = 0; r < 8; ++r) {
        int kk = kt * 32 + hh * 16 + 2 * r;
        union { _Float16 h[2]; unsigned u; } P;
        P.h[0] = (_Float16)W[kk * HC + col];
        P.h[1] = (_Float16)W[(kk + 1) * HC + col];
        pk[((long)tile * 32 + lane) * 8 + r] = P.u;
    }
}

// We is 13x128 (KxN); single k-step, K padded with zeros to 32.
__global__ void k_pack_We(const float* __restrict__ We, unsigned* __restrict__ pk) {
    int nt = blockIdx.x;            // 0..7
    int lane = threadIdx.x;
    int nl = lane & 15, hh = lane >> 4;
    int col = nt * 16 + nl;
    #pragma unroll
    for (int r = 0; r < 8; ++r) {
        int kk = hh * 16 + 2 * r;
        union { _Float16 h[2]; unsigned u; } P;
        P.h[0] = (kk     < ED) ? (_Float16)We[kk * HC + col]       : (_Float16)0.f;
        P.h[1] = (kk + 1 < ED) ? (_Float16)We[(kk + 1) * HC + col] : (_Float16)0.f;
        pk[((long)nt * 32 + lane) * 8 + r] = P.u;
    }
}

// ------------------------------------------------ fused QKV+skip projection
// [q|k|v|x_r] = h16 (Nx128) @ Wpacked (128x512) + bias.
// Block = 8 waves covers 32 rows x 512 cols; wave = (row tile, output matrix).
// A tile staged in LDS once; B frags of each k-step preloaded as one clause.
__global__ __launch_bounds__(256)
void k_gemm_qkvs(const _Float16* __restrict__ h16, const unsigned* __restrict__ pkW,
                 const float* __restrict__ bq, const float* __restrict__ bk,
                 const float* __restrict__ bv, const float* __restrict__ bs,
                 float* __restrict__ q, float* __restrict__ k,
                 float* __restrict__ v, float* __restrict__ xr, int n) {
    __shared__ _Float16 hA[32 * HC];                  // 8 KB activation tile
    const int lane = threadIdx.x & 31;
    const int wave = threadIdx.x >> 5;
    const int rt = wave & 1;               // row tile in block
    const int cg = wave >> 1;              // 0..3 -> which output matrix
    const int row0b = blockIdx.x * 32;
    const int row0 = row0b + rt * 16;
    const int mloc = lane & 15, hh = lane >> 4;

    // stage 32 rows x 128 cols of h16 into LDS (row-clamped)
    for (int i = threadIdx.x; i < 512; i += 256) {    // 512 x 16B chunks
        int r = i >> 4, cchunk = i & 15;
        int grow = row0b + r; if (grow >= n) grow = n - 1;
        ((uint4*)hA)[i] = ((const uint4*)(h16 + (long)grow * HC))[cchunk];
    }
    __syncthreads();

    const int arow_l = rt * 16 + mloc;                // A row within LDS tile
    v8f acc[8] = {};
    for (int ks = 0; ks < 4; ++ks) {
        // A fragment: two contiguous 16B LDS reads (k = ks*32 + hh*16 + g*8 ..)
        AFrag A;
        A.q[0] = *(const uint4*)(hA + arow_l * HC + ks * 32 + hh * 16);
        A.q[1] = *(const uint4*)(hA + arow_l * HC + ks * 32 + hh * 16 + 8);
        // preload all 8 B frags of this k-step in one clause
        AFrag B[8];
        #pragma unroll
        for (int t = 0; t < 8; ++t) {
            const uint4* bp = (const uint4*)(pkW + ((long)(ks * 32 + cg * 8 + t) * 32 + lane) * 8);
            B[t].q[0] = bp[0];
            B[t].q[1] = bp[1];
        }
        #pragma unroll
        for (int t = 0; t < 8; ++t)
            acc[t] = __builtin_amdgcn_wmma_f32_16x16x32_f16(
                false, A.v, false, B[t].v, (short)0, acc[t], false, false);
    }

    float* outp; const float* bias;
    if      (cg == 0) { outp = q;  bias = bq; }
    else if (cg == 1) { outp = k;  bias = bk; }
    else if (cg == 2) { outp = v;  bias = bv; }
    else              { outp = xr; bias = bs; }
    #pragma unroll
    for (int t = 0; t < 8; ++t) {
        int col = t * 16 + mloc;
        float b = bias[col];
        #pragma unroll
        for (int r = 0; r < 8; ++r) {
            int grow = row0 + hh * 8 + r;
            if (grow < n) outp[(long)grow * HC + col] = acc[t][r] + b;
        }
    }
}

// ------------------------------------------------------------- edge embed
// e16 = edge_attr (Ex13) @ We (13x128), K zero-padded to one 32-step.
// B frags stay register-resident; each wave sweeps 8 row tiles (128 edges).
__global__ __launch_bounds__(256)
void k_gemm_edge(const float* __restrict__ ea, const unsigned* __restrict__ pkWe,
                 _Float16* __restrict__ e16, int E) {
    const int lane = threadIdx.x & 31;
    const int wave = threadIdx.x >> 5;
    const int mloc = lane & 15, hh = lane >> 4;
    const int wave_id = blockIdx.x * 8 + wave;

    AFrag B[8];
    #pragma unroll
    for (int t = 0; t < 8; ++t) {
        const uint4* bp = (const uint4*)(pkWe + ((long)t * 32 + lane) * 8);
        B[t].q[0] = bp[0];
        B[t].q[1] = bp[1];
    }

    for (int it = 0; it < 8; ++it) {
        int row0 = (wave_id * 8 + it) * 16;
        int arow = row0 + mloc; if (arow >= E) arow = E - 1;
        AFrag A;
        #pragma unroll
        for (int r = 0; r < 8; ++r) {
            int g = r >> 2, w = r & 3;
            int kk = hh * 16 + r * 2;
            A.h[2 * r]     = (kk     < ED) ? (_Float16)ea[(long)arow * ED + kk]     : (_Float16)0.f;
            A.h[2 * r + 1] = (kk + 1 < ED) ? (_Float16)ea[(long)arow * ED + kk + 1] : (_Float16)0.f;
        }
        #pragma unroll
        for (int t = 0; t < 8; ++t) {
            v8f acc = {};
            acc = __builtin_amdgcn_wmma_f32_16x16x32_f16(
                false, A.v, false, B[t].v, (short)0, acc, false, false);
            int col = t * 16 + mloc;
            #pragma unroll
            for (int r = 0; r < 8; ++r) {
                int grow = row0 + hh * 8 + r;
                if (grow < E) e16[(long)grow * HC + col] = (_Float16)acc[r];
            }
        }
    }
}

// -------------------------------------------------------- scatter softmax
__device__ __forceinline__ unsigned enc_f32(float f) {
    unsigned b = __float_as_uint(f);
    return b ^ ((unsigned)((int)b >> 31) | 0x80000000u);   // order-preserving
}
__device__ __forceinline__ float dec_f32(unsigned e) {
    unsigned b = (e & 0x80000000u) ? (e & 0x7FFFFFFFu) : ~e;
    return __uint_as_float(b);
}

__global__ __launch_bounds__(256)
void k_logits(const float* __restrict__ q, const float* __restrict__ kk,
              const _Float16* __restrict__ e16, const int* __restrict__ ei,
              float* __restrict__ logit, unsigned* __restrict__ mEnc, int E) {
    long idx = (long)blockIdx.x * blockDim.x + threadIdx.x;
    if (idx >= (long)E * NH) return;
    int ed = (int)(idx >> 2), h = (int)(idx & 3);
    int src = ei[ed], dst = ei[E + ed];
    const float*    qp = q   + (long)dst * HC + h * CH;
    const float*    kp = kk  + (long)src * HC + h * CH;
    const _Float16* ep = e16 + (long)ed  * HC + h * CH;
    float s = 0.f;
    #pragma unroll
    for (int i = 0; i < CH; ++i) s += qp[i] * (kp[i] + (float)ep[i]);
    s *= 0.17677669529663687f;           // 1/sqrt(32)
    logit[idx] = s;
    atomicMax(mEnc + (long)dst * NH + h, enc_f32(s));
}

__global__ __launch_bounds__(256)
void k_expsum(float* __restrict__ logit, const unsigned* __restrict__ mEnc,
              const int* __restrict__ ei, float* __restrict__ denom, int E) {
    long idx = (long)blockIdx.x * blockDim.x + threadIdx.x;
    if (idx >= (long)E * NH) return;
    int ed = (int)(idx >> 2), h = (int)(idx & 3);
    int dst = ei[E + ed];
    float m = dec_f32(mEnc[(long)dst * NH + h]);
    float a = __expf(logit[idx] - m);
    logit[idx] = a;                       // reuse buffer: now holds exp term
    atomicAdd(denom + (long)dst * NH + h, a);
}

__global__ __launch_bounds__(256)
void k_messages(const float* __restrict__ aBuf, const float* __restrict__ denom,
                const float* __restrict__ v, const _Float16* __restrict__ e16,
                const int* __restrict__ ei, float* __restrict__ out, int E) {
    long idx = (long)blockIdx.x * blockDim.x + threadIdx.x;
    if (idx >= (long)E * NH) return;
    int ed = (int)(idx >> 2), h = (int)(idx & 3);
    int src = ei[ed], dst = ei[E + ed];
    float alpha = aBuf[idx] / (denom[(long)dst * NH + h] + 1e-16f);
    const float*    vp = v   + (long)src * HC + h * CH;
    const _Float16* ep = e16 + (long)ed  * HC + h * CH;
    float* op = out + (long)dst * HC + h * CH;
    #pragma unroll
    for (int i = 0; i < CH; ++i)
        atomicAdd(op + i, (vp[i] + (float)ep[i]) * alpha);
}

// ----------------------------------------------- beta gate + BN + activation
__global__ __launch_bounds__(128)
void k_beta(const float* __restrict__ out, const float* __restrict__ xr,
            const float* __restrict__ Wb, float* __restrict__ hp, int n) {
    __shared__ float red[128];
    int node = blockIdx.x, c = threadIdx.x;
    float o = out[(long)node * HC + c];
    float x = xr [(long)node * HC + c];
    red[c] = o * Wb[c] + x * Wb[HC + c] + (o - x) * Wb[2 * HC + c];
    __syncthreads();
    for (int s = 64; s > 0; s >>= 1) {
        if (c < s) red[c] += red[c + s];
        __syncthreads();
    }
    float b = 1.f / (1.f + __expf(-red[0]));
    hp[(long)node * HC + c] = b * x + (1.f - b) * o;
}

__global__ __launch_bounds__(256)
void k_bnstats(const float* __restrict__ hp, float* __restrict__ stats, int n) {
    int c = blockIdx.x;                   // one channel per block
    float s = 0.f, ss = 0.f;
    for (int i = threadIdx.x; i < n; i += 256) {
        float v = hp[(long)i * HC + c];
        s += v; ss += v * v;
    }
    __shared__ float rs[256], rq[256];
    rs[threadIdx.x] = s; rq[threadIdx.x] = ss;
    __syncthreads();
    for (int st = 128; st > 0; st >>= 1) {
        if (threadIdx.x < st) { rs[threadIdx.x] += rs[threadIdx.x + st];
                                rq[threadIdx.x] += rq[threadIdx.x + st]; }
        __syncthreads();
    }
    if (threadIdx.x == 0) { stats[c] = rs[0]; stats[HC + c] = rq[0]; }
}

__global__ __launch_bounds__(256)
void k_bn_apply(const float* __restrict__ hp, const float* __restrict__ stats,
                const float* __restrict__ gamma, const float* __restrict__ beta,
                float* __restrict__ hn, int n) {
    long idx = (long)blockIdx.x * blockDim.x + threadIdx.x;
    if (idx >= (long)n * HC) return;
    int c = (int)(idx & (HC - 1));
    float inv_n = 1.f / (float)n;
    float mu = stats[c] * inv_n;
    float var = stats[HC + c] * inv_n - mu * mu;
    float y = (hp[idx] - mu) * rsqrtf(var + 1e-5f) * gamma[c] + beta[c];
    hn[idx] = fmaxf(y, 0.f);
}

// ------------------------------------------------------------- pool + head
__global__ __launch_bounds__(256)
void k_pool(const float* __restrict__ h, const int* __restrict__ batch,
            float* __restrict__ psum, float* __restrict__ pcnt, int n) {
    long idx = (long)blockIdx.x * blockDim.x + threadIdx.x;
    if (idx >= (long)n * HC) return;
    int c = (int)(idx & (HC - 1));
    int node = (int)(idx >> 7);
    int g = batch[node];
    atomicAdd(psum + (long)g * HC + c, h[idx]);
    if (c == 0) atomicAdd(pcnt + g, 1.f);
}

__global__ __launch_bounds__(32)
void k_head(const float* __restrict__ psum, const float* __restrict__ pcnt,
            const float* __restrict__ w1, const float* __restrict__ b1,
            const float* __restrict__ w2, const float* __restrict__ b2,
            float* __restrict__ outp) {
    int g = blockIdx.x, t = threadIdx.x;
    __shared__ float gs[HC];
    __shared__ float hid[CH];
    float cnt = fmaxf(pcnt[g], 1.f);
    for (int i = t; i < HC; i += 32) gs[i] = psum[(long)g * HC + i] / cnt;
    __syncthreads();
    float s = b1[t];
    for (int kk = 0; kk < HC; ++kk) s += gs[kk] * w1[kk * CH + t];
    hid[t] = fmaxf(s, 0.f);
    __syncthreads();
    if (t == 0) {
        float r = b2[0];
        for (int i = 0; i < CH; ++i) r += hid[i] * w2[i];
        outp[g] = r;
    }
}

// --------------------------------------------------------------- launcher
extern "C" void kernel_launch(void* const* d_in, const int* in_sizes, int n_in,
                              void* d_out, int out_size, void* d_ws, size_t ws_size,
                              hipStream_t stream) {
    const float* x   = (const float*)d_in[0];
    const int*   ei  = (const int*)  d_in[1];
    const float* ea  = (const float*)d_in[2];
    const int*   bat = (const int*)  d_in[3];
    // params: layer l base = 4 + 12*l :
    //  +0 Wq +1 bq +2 Wk +3 bk +4 Wv +5 bv +6 We +7 Wskip +8 bskip +9 Wbeta +10 gamma +11 beta
    const float* lin1w = (const float*)d_in[28];
    const float* lin1b = (const float*)d_in[29];
    const float* lin2w = (const float*)d_in[30];
    const float* lin2b = (const float*)d_in[31];

    // ---- workspace carving (256B aligned)
    char* w = (char*)d_ws;
    auto alloc = [&](size_t bytes) -> void* {
        void* p = (void*)w;
        w += (bytes + 255) & ~(size_t)255;
        return p;
    };
    _Float16* h16   = (_Float16*)alloc((size_t)NN * HC * 2);
    unsigned* pkW   = (unsigned*)alloc((size_t)128 * 32 * 8 * 4);
    unsigned* pkWe  = (unsigned*)alloc((size_t)8   * 32 * 8 * 4);
    float*    qb    = (float*)alloc((size_t)NN * HC * 4);
    float*    kb    = (float*)alloc((size_t)NN * HC * 4);
    float*    vb    = (float*)alloc((size_t)NN * HC * 4);
    float*    xrb   = (float*)alloc((size_t)NN * HC * 4);
    _Float16* e16   = (_Float16*)alloc((size_t)EE * HC * 2);
    float*    lg    = (float*)alloc((size_t)EE * NH * 4);
    unsigned* mEnc  = (unsigned*)alloc((size_t)NN * NH * 4);
    float*    denom = (float*)alloc((size_t)NN * NH * 4);
    float*    outb  = (float*)alloc((size_t)NN * HC * 4);
    float*    hp    = (float*)alloc((size_t)NN * HC * 4);
    float*    stats = (float*)alloc((size_t)2 * HC * 4);
    float*    hbuf  = (float*)alloc((size_t)NN * HC * 4);
    float*    psum  = (float*)alloc((size_t)GG * HC * 4);
    float*    pcnt  = (float*)alloc((size_t)GG * 4);

    const long nh_tot  = (long)NN * HC;            // 6.4M
    const long eh_tot  = (long)EE * NH;            // 1.6M
    const int  gNH     = (int)((nh_tot + 255) / 256);     // 25000
    const int  gEH     = (int)((eh_tot + 255) / 256);     // 6250
    const int  gRows   = (NN + 31) / 32;                  // 1563
    const int  gEdge   = (EE / 16 + 63) / 64;             // 391 (8 waves x 8 tiles)

    for (int l = 0; l < 2; ++l) {
        int b = 4 + 12 * l;
        const float* Wq = (const float*)d_in[b + 0];
        const float* bq = (const float*)d_in[b + 1];
        const float* Wk = (const float*)d_in[b + 2];
        const float* bk = (const float*)d_in[b + 3];
        const float* Wv = (const float*)d_in[b + 4];
        const float* bv = (const float*)d_in[b + 5];
        const float* We = (const float*)d_in[b + 6];
        const float* Ws = (const float*)d_in[b + 7];
        const float* bs = (const float*)d_in[b + 8];
        const float* Wb = (const float*)d_in[b + 9];
        const float* gm = (const float*)d_in[b + 10];
        const float* bt = (const float*)d_in[b + 11];
        const float* hin = (l == 0) ? x : hbuf;

        k_cvt_f16 <<<gNH, 256, 0, stream>>>(hin, h16, nh_tot);
        k_pack_qkvs<<<128, 32, 0, stream>>>(Wq, Wk, Wv, Ws, pkW);
        k_gemm_qkvs<<<gRows, 256, 0, stream>>>(h16, pkW, bq, bk, bv, bs,
                                               qb, kb, vb, xrb, NN);
        k_pack_We <<<8, 32, 0, stream>>>(We, pkWe);
        k_gemm_edge<<<gEdge, 256, 0, stream>>>(ea, pkWe, e16, EE);

        k_zero<<<gNH, 256, 0, stream>>>(outb, nh_tot);
        k_zero<<<(NN * NH + 255) / 256, 256, 0, stream>>>((float*)mEnc, (long)NN * NH);
        k_zero<<<(NN * NH + 255) / 256, 256, 0, stream>>>(denom, (long)NN * NH);

        k_logits  <<<gEH, 256, 0, stream>>>(qb, kb, e16, ei, lg, mEnc, EE);
        k_expsum  <<<gEH, 256, 0, stream>>>(lg, mEnc, ei, denom, EE);
        k_messages<<<gEH, 256, 0, stream>>>(lg, denom, vb, e16, ei, outb, EE);

        k_beta    <<<NN, 128, 0, stream>>>(outb, xrb, Wb, hp, NN);
        k_bnstats <<<HC, 256, 0, stream>>>(hp, stats, NN);
        k_bn_apply<<<gNH, 256, 0, stream>>>(hp, stats, gm, bt, hbuf, NN);
    }

    k_zero<<<(GG * HC + 255) / 256, 256, 0, stream>>>(psum, (long)GG * HC);
    k_zero<<<1, 256, 0, stream>>>(pcnt, (long)GG);
    k_pool<<<gNH, 256, 0, stream>>>(hbuf, bat, psum, pcnt, NN);
    k_head<<<GG, 32, 0, stream>>>(psum, pcnt, lin1w, lin1b, lin2w, lin2b, (float*)d_out);
}
